// Encoder_44444321579484
// MI455X (gfx1250) — matrix operand order repaired and checked by reading, not compiled, not run on hardware
//
#include <hip/hip_runtime.h>

// Bidirectional masked GRU (Keras reset_after semantics) for MI455X (gfx1250).
// bf16 WMMA (v_wmma_f32_16x16x32_bf16) for both the input-projection GEMM and
// the per-step recurrent GEMMs; f32 state carry; 128 graph-captured step
// launches with both directions fused per launch. Step kernel spread across
// 128 WGs (512 waves) to shorten the latency-bound sequential critical path.

#define DEV __device__ __forceinline__

typedef __attribute__((ext_vector_type(16))) __bf16 v16bf;
typedef __attribute__((ext_vector_type(8)))  float  v8f;
typedef __attribute__((ext_vector_type(16))) unsigned short v16u;
typedef __attribute__((ext_vector_type(8)))  unsigned short v8u;

constexpr int BATCH = 64;
constexpr int TLEN  = 128;
constexpr int UNITS = 1024;
constexpr int G3    = 3 * UNITS;      // 3072
constexpr int BT    = BATCH * TLEN;   // 8192

DEV unsigned short f2bf(float f) {
  unsigned u = __float_as_uint(f);
  u += 0x7fffu + ((u >> 16) & 1u);    // round-to-nearest-even
  return (unsigned short)(u >> 16);
}

DEV v16bf mkfrag(v8u c0, v8u c1) {
  v16u u = __builtin_shufflevector(c0, c1, 0,1,2,3,4,5,6,7,8,9,10,11,12,13,14,15);
  union { v16u u; v16bf b; } cv; cv.u = u; return cv.b;
}

// A-matrix 16x32 bf16 fragment (ISA 7.12.2): lane half h holds
// K = [8h, 8h+8) and [16+8h, 16+8h+8) of its row (row-major source).
DEV v16bf load_a(const unsigned short* __restrict__ row, int k, int half) {
  v8u c0 = *(const v8u*)(row + k + 8 * half);
  v8u c1 = *(const v8u*)(row + k + 16 + 8 * half);
  return mkfrag(c0, c1);
}

// B-matrix 32x16 bf16 fragment: lane half h holds K = [8h, 8h+8) and
// [16+8h, 16+8h+8) of its column; source is pre-transposed to [N][K] row-major.
DEV v16bf load_b(const unsigned short* __restrict__ row, int k, int half) {
  v8u c0 = *(const v8u*)(row + k + 8 * half);
  v8u c1 = *(const v8u*)(row + k + 16 + 8 * half);
  return mkfrag(c0, c1);
}

DEV v8f wmma_bf16(v16bf a, v16bf b, v8f c) {
  return __builtin_amdgcn_wmma_f32_16x16x32_bf16(false, a, false, b,
                                                 (short)0, c, false, false);
}

DEV float sigmoidf(float x) { return 1.0f / (1.0f + __expf(-x)); }

// -------------------------------------------------------------------------
__global__ void k_zero_u32(unsigned* __restrict__ p, long n) {
  long i  = (long)blockIdx.x * blockDim.x + threadIdx.x;
  long st = (long)gridDim.x * blockDim.x;
  for (; i < n; i += st) p[i] = 0u;
}

// float [UNITS][G3] -> bf16 [G3][UNITS] transpose+convert
__global__ void k_tconv(const float* __restrict__ in, unsigned short* __restrict__ o) {
  long n  = (long)G3 * UNITS;
  long i  = (long)blockIdx.x * blockDim.x + threadIdx.x;
  long st = (long)gridDim.x * blockDim.x;
  for (; i < n; i += st) {
    int k  = (int)(i & (UNITS - 1));
    int nr = (int)(i >> 10);
    o[i] = f2bf(in[(size_t)k * G3 + nr]);
  }
}

// emb gather -> bf16 [BT][UNITS]
__global__ void k_gather(const int* __restrict__ tok, const float* __restrict__ tbl,
                         unsigned short* __restrict__ o) {
  long n  = (long)BT * UNITS;
  long i  = (long)blockIdx.x * blockDim.x + threadIdx.x;
  long st = (long)gridDim.x * blockDim.x;
  for (; i < n; i += st) {
    int row = (int)(i >> 10);
    int k   = (int)(i & (UNITS - 1));
    o[i] = f2bf(tbl[(size_t)tok[row] * UNITS + k]);
  }
}

// xp[dir] = emb_bf16 @ W_dir + b_dir[0]    (A:[BT][U] bf16, B:[G3][U] bf16 (NT))
// grid (BT/64, G3/64, 2), 256 threads = 8 waves; wave w -> M-subtile w&3,
// N-subtiles {2*(w>>2), 2*(w>>2)+1}.
__global__ void __launch_bounds__(256)
k_xp_gemm(const unsigned short* __restrict__ A, const unsigned short* __restrict__ Wt,
          const float* __restrict__ bfp, const float* __restrict__ bbp,
          float* __restrict__ xp) {
  int dir = blockIdx.z;
  const unsigned short* Bm = Wt + (size_t)dir * G3 * UNITS;
  const float* b0 = dir ? bbp : bfp;
  float* o = xp + (size_t)dir * BT * G3;

  int m0 = blockIdx.x * 64, n0 = blockIdx.y * 64;
  int w = threadIdx.x >> 5, lane = threadIdx.x & 31;
  int half = lane >> 4, l16 = lane & 15;
  int mi = w & 3, nj0 = (w >> 2) * 2;

  const unsigned short* ar  = A  + (size_t)(m0 + mi * 16 + l16) * UNITS;
  const unsigned short* br0 = Bm + (size_t)(n0 + nj0 * 16 + l16) * UNITS;
  const unsigned short* br1 = br0 + (size_t)16 * UNITS;

  v8f acc0 = {}, acc1 = {};
  for (int k = 0; k < UNITS; k += 32) {
    v16bf a = load_a(ar, k, half);
    acc0 = wmma_bf16(a, load_b(br0, k, half), acc0);
    acc1 = wmma_bf16(a, load_b(br1, k, half), acc1);
  }
  int m  = m0 + mi * 16 + half * 8;          // C/D layout: M = 8*(lane>=16)+j
  int na = n0 + nj0 * 16 + l16;              //            N = lane&15
  float bz0 = b0[na], bz1 = b0[na + 16];
#pragma unroll
  for (int j = 0; j < 8; ++j) {
    o[(size_t)(m + j) * G3 + na]      = acc0[j] + bz0;
    o[(size_t)(m + j) * G3 + na + 16] = acc1[j] + bz1;
  }
}

// One recurrent step, both directions (blockIdx.y). grid (UNITS/16, 2),
// 128 threads = 4 waves. Each WG owns a [64 batch x 16 unit] output slice;
// wave w owns batch rows [16w,16w+16) and carries 3 accumulators (z,r,h).
// Per K-iter: 1 A-frag load + 3 B-frag loads + 3 WMMAs -> short critical path,
// 128 WGs per step to cover the latency-bound recurrence with many waves.
__global__ void __launch_bounds__(128)
k_gru_step(const float* __restrict__ xp, const unsigned short* __restrict__ Ut,
           const float* __restrict__ bfp, const float* __restrict__ bbp,
           const int* __restrict__ tok,
           float* __restrict__ hbuf, unsigned short* __restrict__ hbf,
           float* __restrict__ pout, float* __restrict__ out, int s) {
  int dir = blockIdx.y;
  int t = dir ? (TLEN - 1 - s) : s;
  const float*          xpd = xp + (size_t)dir * BT * G3;
  const unsigned short* Ud  = Ut + (size_t)dir * G3 * UNITS;
  const float*          b1  = (dir ? bbp : bfp) + G3;
  const float*    h_in   = hbuf + ((size_t)dir * 2 + (s & 1)) * BATCH * UNITS;
  float*          h_out  = hbuf + ((size_t)dir * 2 + ((s + 1) & 1)) * BATCH * UNITS;
  const unsigned short* hb_in  = hbf + ((size_t)dir * 2 + (s & 1)) * BATCH * UNITS;
  unsigned short*       hb_out = hbf + ((size_t)dir * 2 + ((s + 1) & 1)) * BATCH * UNITS;
  float* po = pout + (size_t)dir * BATCH * UNITS;

  int n0 = blockIdx.x * 16;
  int w = threadIdx.x >> 5, lane = threadIdx.x & 31;
  int half = lane >> 4, l16 = lane & 15;
  int mi = w;                                  // 4 waves -> 4 batch subtiles

  const unsigned short* ar = hb_in + (size_t)(mi * 16 + l16) * UNITS;
  const unsigned short* br[3];
#pragma unroll
  for (int g = 0; g < 3; ++g)
    br[g] = Ud + (size_t)(g * UNITS + n0 + l16) * UNITS;

  v8f acc[3] = {};
  for (int k = 0; k < UNITS; k += 32) {
    v16bf a = load_a(ar, k, half);
#pragma unroll
    for (int g = 0; g < 3; ++g)
      acc[g] = wmma_bf16(a, load_b(br[g], k, half), acc[g]);
  }

  int n = n0 + l16;
  float b1z = b1[n], b1r = b1[n + UNITS], b1h = b1[n + 2 * UNITS];
#pragma unroll
  for (int j = 0; j < 8; ++j) {
    int b_ = mi * 16 + half * 8 + j;                         // batch row
    float h_old = h_in[(size_t)b_ * UNITS + n];
    const float* xrow = xpd + ((size_t)b_ * TLEN + t) * G3;
    float z  = sigmoidf(xrow[n]             + acc[0][j] + b1z);
    float r  = sigmoidf(xrow[n + UNITS]     + acc[1][j] + b1r);
    float hh = tanhf   (xrow[n + 2 * UNITS] + r * (acc[2][j] + b1h));
    float hnew = z * h_old + (1.0f - z) * hh;
    bool  msk = tok[b_ * TLEN + t] != 0;
    float ov    = msk ? hnew : po[(size_t)b_ * UNITS + n];
    float hnext = msk ? hnew : h_old;
    out[((size_t)b_ * TLEN + t) * UNITS + n] += ov;          // fwd+bwd fuse
    po[(size_t)b_ * UNITS + n]     = ov;
    h_out[(size_t)b_ * UNITS + n]  = hnext;
    hb_out[(size_t)b_ * UNITS + n] = f2bf(hnext);
  }
}

// -------------------------------------------------------------------------
extern "C" void kernel_launch(void* const* d_in, const int* in_sizes, int n_in,
                              void* d_out, int out_size, void* d_ws, size_t ws_size,
                              hipStream_t stream) {
  (void)in_sizes; (void)n_in; (void)out_size; (void)ws_size;
  const int*   tokens = (const int*)  d_in[0];
  const float* embtbl = (const float*)d_in[1];
  const float* Wf = (const float*)d_in[2];
  const float* Uf = (const float*)d_in[3];
  const float* bf = (const float*)d_in[4];
  const float* Wb = (const float*)d_in[5];
  const float* Ub = (const float*)d_in[6];
  const float* bb = (const float*)d_in[7];
  float* out = (float*)d_out;

  // Workspace layout (~246 MB)
  char* p = (char*)d_ws;
  float* xp = (float*)p;                    p += (size_t)2 * BT * G3 * sizeof(float);
  unsigned short* Wt = (unsigned short*)p;  p += (size_t)2 * G3 * UNITS * 2;
  unsigned short* Ut = (unsigned short*)p;  p += (size_t)2 * G3 * UNITS * 2;
  unsigned short* embb = (unsigned short*)p; p += (size_t)BT * UNITS * 2;
  float* hbuf = (float*)p;                  p += (size_t)4 * BATCH * UNITS * sizeof(float);
  float* pout = (float*)p;                  p += (size_t)2 * BATCH * UNITS * sizeof(float);
  unsigned short* hbf = (unsigned short*)p; p += (size_t)4 * BATCH * UNITS * 2;

  // Zero output (we accumulate fwd+bwd into it) and all carried state.
  k_zero_u32<<<2048, 256, 0, stream>>>((unsigned*)out, (long)BT * UNITS);
  long state_u32 = ((long)6 * BATCH * UNITS * 4 + (long)4 * BATCH * UNITS * 2) / 4;
  k_zero_u32<<<256, 256, 0, stream>>>((unsigned*)hbuf, state_u32);

  // Weight transpose+convert to bf16 [N][K]; embedding gather to bf16.
  k_tconv<<<2048, 256, 0, stream>>>(Wf, Wt);
  k_tconv<<<2048, 256, 0, stream>>>(Wb, Wt + (size_t)G3 * UNITS);
  k_tconv<<<2048, 256, 0, stream>>>(Uf, Ut);
  k_tconv<<<2048, 256, 0, stream>>>(Ub, Ut + (size_t)G3 * UNITS);
  k_gather<<<2048, 256, 0, stream>>>(tokens, embtbl, embb);

  // Input projection for both directions: [8192x1024] @ [1024x3072] (+bias).
  dim3 gg(BT / 64, G3 / 64, 2);
  k_xp_gemm<<<gg, 256, 0, stream>>>(embb, Wt, bf, bb, xp);

  // Sequential recurrence, both directions per launch, 128 WGs per step.
  dim3 gs(UNITS / 16, 2);
  for (int s = 0; s < TLEN; ++s)
    k_gru_step<<<gs, 128, 0, stream>>>(xp, Ut, bf, bb, tokens,
                                       hbuf, hbf, pout, out, s);
}
